// Mamba1D_67568425501009
// MI455X (gfx1250) — matrix operand order repaired and checked by reading, not compiled, not run on hardware
//
#include <hip/hip_runtime.h>
#include <hip/hip_bf16.h>

// ---------------------------------------------------------------------------
// Model dims (compile-time)
// ---------------------------------------------------------------------------
#define BB   8
#define CC   32
#define TT   8192
#define DM   256
#define NN   16
#define KK   4
#define LL   6
#define CD   64
#define DI   512           // DM * EX
#define MM   (BB * TT)     // 65536 rows
#define XPN  48            // 2N+1 = 33 padded to 48 (3 WMMA n-tiles)
#define BROWS 128          // B rows staged per block (always 128, padded)

typedef float v8f  __attribute__((ext_vector_type(8)));
typedef __bf16 bf16x16 __attribute__((ext_vector_type(16)));
typedef __bf16 bf16x8  __attribute__((ext_vector_type(8)));

// Async global->LDS (CDNA5): guarded, with sync fallback
#if defined(__has_builtin)
#if __has_builtin(__builtin_amdgcn_global_load_async_to_lds_b128) && \
    __has_builtin(__builtin_amdgcn_s_wait_asynccnt)
#define HAVE_ASYNC_LDS 1
#endif
#endif
#ifndef HAVE_ASYNC_LDS
#define HAVE_ASYNC_LDS 0
#endif

// Builtin expects: (int __vector(4) AS1*, int __vector(4) AS3*, imm, imm)
typedef int v4i __attribute__((vector_size(16)));
typedef __attribute__((address_space(1))) v4i gbl_v4i;
typedef __attribute__((address_space(3))) v4i lds_v4i;

// Integer-detour address space casts (numerically correct on AMDGPU:
// flat global == AS1; flat LDS addr[31:0] == LDS offset per ISA 10.2)
__device__ __forceinline__ gbl_v4i* to_gbl_v4i(const void* p) {
    return (gbl_v4i*)(unsigned long long)(uintptr_t)p;
}
__device__ __forceinline__ lds_v4i* to_lds_v4i(void* p) {
    return (lds_v4i*)(unsigned)(uintptr_t)p;
}

// ---------------------------------------------------------------------------
// bf16 helpers (bit-level, round-to-nearest-even)
// ---------------------------------------------------------------------------
__device__ __forceinline__ unsigned short f2bf(float f) {
    union { float f; unsigned u; } v; v.f = f;
    unsigned u = v.u;
    u += 0x7fffu + ((u >> 16) & 1u);
    return (unsigned short)(u >> 16);
}
__device__ __forceinline__ float bf2f(unsigned short s) {
    union { unsigned u; float f; } v; v.u = ((unsigned)s) << 16;
    return v.f;
}
__device__ __forceinline__ float siluf(float x) {
    return x * (1.0f / (1.0f + __expf(-x)));
}

// ---------------------------------------------------------------------------
// Weight cast+transpose: src [K, Nout] f32 -> dst [NoutPad, K] bf16 (rows = n)
// ---------------------------------------------------------------------------
__global__ __launch_bounds__(256) void wcast_kernel(
    const float* __restrict__ src, unsigned short* __restrict__ dst,
    int K, int Nout, int NoutPad)
{
    int idx = blockIdx.x * 256 + threadIdx.x;
    int total = NoutPad * K;
    if (idx >= total) return;
    int n = idx / K, k = idx - n * K;
    dst[idx] = (n < Nout) ? f2bf(src[k * Nout + n]) : (unsigned short)0;
}

// ---------------------------------------------------------------------------
// Input embedding: x[bt,d] = sum_c ob[b,c,t]*inW[d,c] + inB[d]
// ---------------------------------------------------------------------------
__global__ __launch_bounds__(256) void in_embed_kernel(
    const float* __restrict__ ob, const float* __restrict__ inW,
    const float* __restrict__ inB, float* __restrict__ X)
{
    __shared__ float obrow[CC];
    int row = blockIdx.x;                 // bt
    int b = row >> 13, t = row & (TT - 1);
    int d = threadIdx.x;
    if (d < CC) obrow[d] = ob[((size_t)(b * CC + d)) * TT + t];
    __syncthreads();
    float acc = inB[d];
    const float* w = inW + d * CC;
    #pragma unroll
    for (int c = 0; c < CC; ++c) acc = fmaf(obrow[c], w[c], acc);
    X[(size_t)row * DM + d] = acc;
}

// ---------------------------------------------------------------------------
// LayerNorm over DM=256, output bf16.  8 waves/block, 1 wave per row.
// ---------------------------------------------------------------------------
__global__ __launch_bounds__(256) void ln_bf16_kernel(
    const float* __restrict__ X, const float* __restrict__ w,
    const float* __restrict__ bi, unsigned short* __restrict__ Hb)
{
    int wave = threadIdx.x >> 5, lane = threadIdx.x & 31;
    int row = blockIdx.x * 8 + wave;
    const float* xr = X + (size_t)row * DM;
    float v[8], s = 0.f;
    #pragma unroll
    for (int i = 0; i < 8; ++i) { v[i] = xr[lane * 8 + i]; s += v[i]; }
    #pragma unroll
    for (int off = 16; off > 0; off >>= 1) s += __shfl_xor(s, off);
    float m = s * (1.0f / DM);
    float ss = 0.f;
    #pragma unroll
    for (int i = 0; i < 8; ++i) { float d = v[i] - m; ss += d * d; }
    #pragma unroll
    for (int off = 16; off > 0; off >>= 1) ss += __shfl_xor(ss, off);
    float rs = rsqrtf(ss * (1.0f / DM) + 1e-5f);
    unsigned short* o = Hb + (size_t)row * DM;
    #pragma unroll
    for (int i = 0; i < 8; ++i) {
        int c = lane * 8 + i;
        o[c] = f2bf((v[i] - m) * rs * w[c] + bi[c]);
    }
}

// ---------------------------------------------------------------------------
// Tiled bf16 WMMA GEMM:  C[M,Nout] = A[M,K] * Bt^T   (Bt is [NoutPad>=n0+128,K])
// block: 256 threads (8 waves); tile 128(M) x NTILE(N); Ktile = 32.
// Double-buffered LDS, async global->LDS staging, pipelined B fragments.
// MODE 0: plain f32 out | MODE 1: split xi/z | MODE 2: residual + FiLM
// ---------------------------------------------------------------------------
template <int NTILE, int MODE>
__global__ __launch_bounds__(256) void wmma_gemm_kernel(
    const unsigned short* __restrict__ A,   // [M,K] bf16
    const unsigned short* __restrict__ Bt,  // [NoutPad,K] bf16 (>=128 rows/blk)
    int K, int Nout,
    float* __restrict__ outF,
    float* __restrict__ U, unsigned short* __restrict__ Zb,
    const float* __restrict__ res, const float* __restrict__ G,
    const float* __restrict__ Be, float* __restrict__ Xnext)
{
    constexpr int NB = NTILE / 16;
    __shared__ __align__(16) unsigned short As[2][128 * 32];
    __shared__ __align__(16) unsigned short Bs[2][BROWS * 32];

    int tid = threadIdx.x;
    int lane = tid & 31, wave = tid >> 5;
    int m0 = blockIdx.x * 128;
    int n0 = blockIdx.y * NTILE;

    v8f acc[NB];
    #pragma unroll
    for (int nb = 0; nb < NB; ++nb)
        acc[nb] = (v8f){0.f,0.f,0.f,0.f,0.f,0.f,0.f,0.f};

    int lr = tid >> 1;                 // 0..127
    int lh = (tid & 1) << 4;           // 0 or 16 elements
    const unsigned short* gA = A  + (size_t)(m0 + lr) * K + lh;
    const unsigned short* gB = Bt + (size_t)(n0 + lr) * K + lh;

    auto stage = [&](int buf, int k0) {
        const unsigned short* ga = gA + k0;
        const unsigned short* gb = gB + k0;
        unsigned short* la = &As[buf][lr * 32 + lh];
        unsigned short* lb = &Bs[buf][lr * 32 + lh];
#if HAVE_ASYNC_LDS
        __builtin_amdgcn_global_load_async_to_lds_b128(
            to_gbl_v4i(ga), to_lds_v4i(la), 0, 0);
        __builtin_amdgcn_global_load_async_to_lds_b128(
            to_gbl_v4i(gb), to_lds_v4i(lb), 0, 0);
#else
        *(uint4*)la = *(const uint4*)ga;
        *(uint4*)lb = *(const uint4*)gb;
#endif
    };

    // fragment geometry (loop-invariant)
    int mA  = (wave << 4) + (lane & 15);
    int kbA = (lane < 16) ? 0 : 8;
    int nBf = lane & 15;
    int klB = (lane < 16) ? 0 : 8;

    auto loadB = [&](bf16x16& f, int buf, int nb) {
        ((bf16x8*)&f)[0] = *(const bf16x8*)&Bs[buf][(nb * 16 + nBf) * 32 + klB];
        ((bf16x8*)&f)[1] = *(const bf16x8*)&Bs[buf][(nb * 16 + nBf) * 32 + klB + 16];
    };

    // compute on one staged K-tile: pipelined B-fragment rotation so the WMMA
    // on bfv[nb] overlaps the ds_loads of bfv[nb+1] (partial dscnt waits)
    auto compute = [&](int buf) {
        bf16x16 af;
        ((bf16x8*)&af)[0] = *(const bf16x8*)&As[buf][mA * 32 + kbA];
        ((bf16x8*)&af)[1] = *(const bf16x8*)&As[buf][mA * 32 + kbA + 16];
        bf16x16 curf, nxtf;
        loadB(curf, buf, 0);
        #pragma unroll
        for (int nb = 0; nb < NB; ++nb) {
            if (nb + 1 < NB) loadB(nxtf, buf, nb + 1);
            acc[nb] = __builtin_amdgcn_wmma_f32_16x16x32_bf16(
                false, af, false, curf, (short)0, acc[nb], false, false);
            curf = nxtf;
        }
    };

    const int nk = K >> 5;
    stage(0, 0);

    // steady state: stage next tile, wait for current, compute
    for (int i = 0; i < nk - 1; ++i) {
        int buf = i & 1;
        stage(buf ^ 1, (i + 1) << 5);
#if HAVE_ASYNC_LDS
        __builtin_amdgcn_s_wait_asynccnt(2);   // tile i landed; i+1 in flight
#endif
        __syncthreads();
        compute(buf);
        __syncthreads();   // done reading buf before it is restaged
    }
    // epilogue tile
#if HAVE_ASYNC_LDS
    __builtin_amdgcn_s_wait_asynccnt(0);
#endif
    __syncthreads();
    compute((nk - 1) & 1);

    // epilogue: VGPR r holds M = waveM + r (+8 for hi lanes), N = lane%16
    int rowb = m0 + (wave << 4) + ((lane >= 16) ? 8 : 0);
    int colL = lane & 15;
    #pragma unroll
    for (int nb = 0; nb < NB; ++nb) {
        int col = n0 + (nb << 4) + colL;
        #pragma unroll
        for (int r = 0; r < 8; ++r) {
            int row = rowb + r;
            float v = acc[nb][r];
            if constexpr (MODE == 0) {
                outF[(size_t)row * Nout + col] = v;
            } else if constexpr (MODE == 1) {
                if (col < DI) U[(size_t)row * DI + col] = v;
                else          Zb[(size_t)row * DI + (col - DI)] = f2bf(v);
            } else {
                int b = row >> 13;
                float val = v + res[(size_t)row * DM + col];
                Xnext[(size_t)row * DM + col] =
                    G[b * DM + col] * val + Be[b * DM + col];
            }
        }
    }
}

// ---------------------------------------------------------------------------
// Causal depthwise conv (K=4) + SiLU:  V f32 + Ab bf16
// ---------------------------------------------------------------------------
__global__ __launch_bounds__(256) void conv_silu_kernel(
    const float* __restrict__ U, const float* __restrict__ cw,
    const float* __restrict__ cb, float* __restrict__ V,
    unsigned short* __restrict__ Ab)
{
    int idx = blockIdx.x * 256 + threadIdx.x;      // over M*DI
    int d = idx & (DI - 1);
    int row = idx >> 9;                             // bt
    int t = row & (TT - 1);
    float acc = cb[d];
    const float* w = cw + d * KK;
    #pragma unroll
    for (int k = 0; k < KK; ++k) {
        int tt = t - (KK - 1) + k;
        if (tt >= 0)
            acc = fmaf(w[k], U[(size_t)(row - (KK - 1) + k) * DI + d], acc);
    }
    float s = siluf(acc);
    V[idx] = s;
    Ab[idx] = f2bf(s);
}

// ---------------------------------------------------------------------------
// Sequential selective scan. grid = (DI/128, B), 128 threads, 16 states/thread.
// ---------------------------------------------------------------------------
__global__ __launch_bounds__(128) void scan_kernel(
    const float* __restrict__ XD, const float* __restrict__ V,
    const float* __restrict__ A_log, const float* __restrict__ Dp,
    float* __restrict__ U)
{
    int d = blockIdx.x * 128 + threadIdx.x;
    int b = blockIdx.y;
    float An[NN];
    #pragma unroll
    for (int n = 0; n < NN; ++n) An[n] = -__expf(A_log[n]);
    float Dd = Dp[d];
    float h[NN];
    #pragma unroll
    for (int n = 0; n < NN; ++n) h[n] = 0.f;

    size_t base = (size_t)b * TT;
    for (int t = 0; t < TT; ++t) {
        size_t r = base + t;
        const float* xd = XD + r * XPN;
        float dlt = xd[0];
        dlt = (dlt > 20.f) ? dlt : log1pf(__expf(dlt));
        float x = V[r * DI + d];
        float dx = dlt * x;
        float y = 0.f;
        #pragma unroll
        for (int n = 0; n < NN; ++n) {
            float a = __expf(An[n] * dlt);
            h[n] = fmaf(a, h[n], dx * xd[1 + n]);
            y = fmaf(h[n], xd[1 + NN + n], y);
        }
        U[r * DI + d] = fmaf(Dd, x, y);
    }
}

// ---------------------------------------------------------------------------
// Gate: Ab = bf16( y * silu(z) )
// ---------------------------------------------------------------------------
__global__ __launch_bounds__(256) void gate_kernel(
    const float* __restrict__ U, const unsigned short* __restrict__ Zb,
    unsigned short* __restrict__ Ab)
{
    int idx = blockIdx.x * 256 + threadIdx.x;
    float z = bf2f(Zb[idx]);
    Ab[idx] = f2bf(U[idx] * siluf(z));
}

// ---------------------------------------------------------------------------
// Conditioning: g/beta per (b,d) from embed[odor_ids]
// ---------------------------------------------------------------------------
__global__ __launch_bounds__(256) void cond_prep_kernel(
    const float* __restrict__ embed, const int* __restrict__ ids,
    const float* __restrict__ gw, const float* __restrict__ gb,
    const float* __restrict__ bw, const float* __restrict__ bbias,
    float* __restrict__ G, float* __restrict__ Be)
{
    int idx = blockIdx.x * 256 + threadIdx.x;     // B*DM
    int b = idx >> 8, d = idx & (DM - 1);
    const float* c = embed + ids[b] * CD;
    float g = gb[d], be = bbias[d];
    #pragma unroll 8
    for (int cd = 0; cd < CD; ++cd) {
        g  = fmaf(c[cd], gw[cd * DM + d], g);
        be = fmaf(c[cd], bw[cd * DM + d], be);
    }
    G[idx] = g;
    Be[idx] = be;
}

// ---------------------------------------------------------------------------
// Final LN + output projection + residual.
// ---------------------------------------------------------------------------
__global__ __launch_bounds__(256) void final_kernel(
    const float* __restrict__ X, const float* __restrict__ fw,
    const float* __restrict__ fb, const float* __restrict__ outW,
    const float* __restrict__ outB, const float* __restrict__ ob,
    float* __restrict__ out)
{
    __shared__ float xs[8][DM];
    int wave = threadIdx.x >> 5, lane = threadIdx.x & 31;
    int row = blockIdx.x * 8 + wave;
    const float* xr = X + (size_t)row * DM;
    float v[8], s = 0.f;
    #pragma unroll
    for (int i = 0; i < 8; ++i) { v[i] = xr[lane * 8 + i]; s += v[i]; }
    #pragma unroll
    for (int off = 16; off > 0; off >>= 1) s += __shfl_xor(s, off);
    float m = s * (1.0f / DM);
    float ss = 0.f;
    #pragma unroll
    for (int i = 0; i < 8; ++i) { float d = v[i] - m; ss += d * d; }
    #pragma unroll
    for (int off = 16; off > 0; off >>= 1) ss += __shfl_xor(ss, off);
    float rs = rsqrtf(ss * (1.0f / DM) + 1e-5f);
    #pragma unroll
    for (int i = 0; i < 8; ++i) {
        int c = lane * 8 + i;
        xs[wave][c] = (v[i] - m) * rs * fw[c] + fb[c];
    }
    __syncthreads();
    int c = lane;                                 // 0..31 = C channels
    float acc = outB[c];
    const float* w = outW + c * DM;
    #pragma unroll 8
    for (int d = 0; d < DM; ++d) acc = fmaf(xs[wave][d], w[d], acc);
    int b = row >> 13, t = row & (TT - 1);
    size_t o = ((size_t)(b * CC + c)) * TT + t;
    out[o] = ob[o] + acc;
}

// ---------------------------------------------------------------------------
// Host-side launch
// ---------------------------------------------------------------------------
extern "C" void kernel_launch(void* const* d_in, const int* in_sizes, int n_in,
                              void* d_out, int out_size, void* d_ws, size_t ws_size,
                              hipStream_t stream) {
    (void)in_sizes; (void)n_in; (void)out_size; (void)ws_size;
    const float* ob       = (const float*)d_in[0];
    const int*   ids      = (const int*)  d_in[1];
    const float* embed    = (const float*)d_in[2];
    const float* inW      = (const float*)d_in[3];
    const float* inB      = (const float*)d_in[4];
    const float* norm_w   = (const float*)d_in[5];
    const float* norm_b   = (const float*)d_in[6];
    const float* in_proj  = (const float*)d_in[7];
    const float* conv_w   = (const float*)d_in[8];
    const float* conv_b   = (const float*)d_in[9];
    const float* xproj_w  = (const float*)d_in[10];
    const float* A_log    = (const float*)d_in[11];
    const float* Dp       = (const float*)d_in[12];
    const float* out_w    = (const float*)d_in[13];
    const float* gamma_w  = (const float*)d_in[14];
    const float* gamma_b  = (const float*)d_in[15];
    const float* beta_w   = (const float*)d_in[16];
    const float* beta_b   = (const float*)d_in[17];
    const float* fnorm_w  = (const float*)d_in[18];
    const float* fnorm_b  = (const float*)d_in[19];
    const float* outW     = (const float*)d_in[20];
    const float* outB     = (const float*)d_in[21];
    float* outp = (float*)d_out;

    // workspace carve-up (256B aligned slots)
    char* ws = (char*)d_ws;
    size_t off = 0;
    auto take = [&](size_t bytes) -> char* {
        char* p = ws + off;
        off += (bytes + 255) & ~(size_t)255;
        return p;
    };
    unsigned short* wIn  = (unsigned short*)take((size_t)LL * 1024 * DM * 2);
    unsigned short* wXp  = (unsigned short*)take((size_t)LL * BROWS * DI * 2); // padded to 128 rows
    unsigned short* wOut = (unsigned short*)take((size_t)LL * DM * DI * 2);
    float*          X0   = (float*)take((size_t)MM * DM * 4);
    float*          X1   = (float*)take((size_t)MM * DM * 4);
    unsigned short* Hb   = (unsigned short*)take((size_t)MM * DM * 2);
    float*          Ubuf = (float*)take((size_t)MM * DI * 4);
    float*          Vbuf = (float*)take((size_t)MM * DI * 4);
    unsigned short* Zb   = (unsigned short*)take((size_t)MM * DI * 2);
    unsigned short* Ab   = (unsigned short*)take((size_t)MM * DI * 2);
    float*          XD   = (float*)take((size_t)MM * XPN * 4);
    float*          Gb   = (float*)take((size_t)BB * DM * 4);
    float*          Beb  = (float*)take((size_t)BB * DM * 4);

    // 1) weight cast+transpose (per layer)
    for (int l = 0; l < LL; ++l) {
        {   int tot = 1024 * DM;
            wcast_kernel<<<(tot + 255) / 256, 256, 0, stream>>>(
                in_proj + (size_t)l * DM * 1024, wIn + (size_t)l * 1024 * DM,
                DM, 1024, 1024);
        }
        {   int tot = BROWS * DI;   // pad xproj weights to 128 rows (zeros)
            wcast_kernel<<<(tot + 255) / 256, 256, 0, stream>>>(
                xproj_w + (size_t)l * DI * 33, wXp + (size_t)l * BROWS * DI,
                DI, 33, BROWS);
        }
        {   int tot = DM * DI;
            wcast_kernel<<<(tot + 255) / 256, 256, 0, stream>>>(
                out_w + (size_t)l * DI * DM, wOut + (size_t)l * DM * DI,
                DI, DM, DM);
        }
    }

    // 2) input embedding
    in_embed_kernel<<<MM, 256, 0, stream>>>(ob, inW, inB, X0);

    float* cur = X0;
    float* nxt = X1;
    const int EW = (MM * DI) / 256;   // elementwise grid over M*DI

    for (int l = 0; l < LL; ++l) {
        // LN -> bf16
        ln_bf16_kernel<<<MM / 8, 256, 0, stream>>>(
            cur, norm_w + l * DM, norm_b + l * DM, Hb);

        // in_proj GEMM (split epilogue): [M,256] x [256,1024]
        wmma_gemm_kernel<128, 1><<<dim3(MM / 128, 1024 / 128), 256, 0, stream>>>(
            Hb, wIn + (size_t)l * 1024 * DM, DM, 1024,
            nullptr, Ubuf, Zb, nullptr, nullptr, nullptr, nullptr);

        // depthwise conv + SiLU
        conv_silu_kernel<<<EW, 256, 0, stream>>>(
            Ubuf, conv_w + (size_t)l * DI * KK, conv_b + l * DI, Vbuf, Ab);

        // xproj GEMM: [M,512] x [512,48]  (B rows padded to 128 in wXp)
        wmma_gemm_kernel<XPN, 0><<<dim3(MM / 128, 1), 256, 0, stream>>>(
            Ab, wXp + (size_t)l * BROWS * DI, DI, XPN,
            XD, nullptr, nullptr, nullptr, nullptr, nullptr, nullptr);

        // selective scan (sequential in T)
        scan_kernel<<<dim3(DI / 128, BB), 128, 0, stream>>>(
            XD, Vbuf, A_log + l * NN, Dp + l * DI, Ubuf);

        // gate: y * silu(z) -> bf16
        gate_kernel<<<EW, 256, 0, stream>>>(Ubuf, Zb, Ab);

        // conditioning params
        cond_prep_kernel<<<(BB * DM) / 256, 256, 0, stream>>>(
            embed, ids,
            gamma_w + (size_t)l * CD * DM, gamma_b + l * DM,
            beta_w + (size_t)l * CD * DM, beta_b + l * DM, Gb, Beb);

        // out GEMM + residual + FiLM: [M,512] x [512,256]
        wmma_gemm_kernel<128, 2><<<dim3(MM / 128, DM / 128), 256, 0, stream>>>(
            Ab, wOut + (size_t)l * DM * DI, DI, DM,
            nullptr, nullptr, nullptr, cur, Gb, Beb, nxt);

        float* tmp = cur; cur = nxt; nxt = tmp;
    }

    // final LN + output projection + residual
    final_kernel<<<MM / 8, 256, 0, stream>>>(
        cur, fnorm_w, fnorm_b, outW, outB, ob, outp);
}
